// ODEFunc_46428596469879
// MI455X (gfx1250) — matrix operand order repaired and checked by reading, not compiled, not run on hardware
//
#include <hip/hip_runtime.h>
#include <hip/hip_bf16.h>

// ---------------------------------------------------------------------------
// CDNA5 (gfx1250) GNN forward: all dense MLP work on v_wmma_f32_16x16x32_f16.
// Edge pipeline: 32 edges/block, fused (edge|gate) layer-1 -> 160x512 GEMM.
// ---------------------------------------------------------------------------

typedef __attribute__((ext_vector_type(16))) _Float16 v16h;
typedef __attribute__((ext_vector_type(8)))  _Float16 v8h;
typedef __attribute__((ext_vector_type(8)))  float    v8f;

#define FDIM 67          // node feature dim F == message dim M
#define HDIM 256         // hidden
#define KP_EDGE 160      // 2F+8 = 142 padded to 32-mult
#define KP_NODE 160      // F+M  = 134 padded
#define KP_WORLD 96      // F+3  = 70 padded
#define NP_OUT 80        // 67 padded to 16-mult
#define EROWS 32         // edges per block
#define H2 (2 * HDIM)    // fused layer-1 width (edge 0..255 | gate 256..511)

union V16U { v16h v; v8h h[2]; };

// A-fragment (16x32 f16, M x K) from LDS row-major tile.
// ISA 7.12.2: e<8 -> K = hi*8+e ; e>=8 -> K = 16+hi*8+(e-8)
// => two contiguous 8-half (16B) runs per lane -> two ds_load_b128.
__device__ inline v16h load_a_frag(const _Float16* base, int strideH, int kt, int lane) {
    int m = lane & 15, hi = lane >> 4;
    const _Float16* row = base + m * strideH + kt * 32;
    V16U u;
    u.h[0] = *(const v8h*)(row + hi * 8);
    u.h[1] = *(const v8h*)(row + 16 + hi * 8);
    return u.v;
}

// B-fragment: weights pre-packed so each lane's 16 halves are contiguous.
__device__ inline v16h load_b_frag(const _Float16* pk, int KT, int nt, int kt, int lane) {
    return *(const v16h*)(pk + (((size_t)nt * KT + kt) * 32 + lane) * 16);
}

__device__ inline v8f wmma_f16(v16h a, v16h b, v8f c) {
    return __builtin_amdgcn_wmma_f32_16x16x32_f16(
        /*neg_a=*/false, a, /*neg_b=*/false, b,
        /*c_mod=*/(short)0, c, /*reuse_a=*/false, /*reuse_b=*/false);
}

// Layer 1 (16-row blocks): X[16 x KT*32] @ W1 -> H[16x256] f16, bias+ReLU.
__device__ inline void mlp_layer1(const _Float16* sX, int strideX, int KT,
                                  const _Float16* W1, const float* b1,
                                  _Float16* sH, int lane, int wave) {
    int ln = lane & 15, hi = lane >> 4;
    for (int j = 0; j < 2; ++j) {
        int nt = wave * 2 + j;
        v8f acc = {};
        for (int kt = 0; kt < KT; ++kt) {
            v16h a = load_a_frag(sX, strideX, kt, lane);
            v16h b = load_b_frag(W1, KT, nt, kt, lane);
            acc = wmma_f16(a, b, acc);
        }
        int n = nt * 16 + ln;
        float bias = b1[n];
        for (int r = 0; r < 8; ++r) {
            float v = acc[r] + bias;
            sH[(hi * 8 + r) * HDIM + n] = (_Float16)fmaxf(v, 0.0f);
        }
    }
}

// Layer 2 (16-row blocks): H[16x256] @ W2 -> sOut[16 x strideOut] f32, +bias.
__device__ inline void mlp_layer2(const _Float16* sH, const _Float16* W2, const float* b2,
                                  float* sOut, int strideOut, int NT, int Nout,
                                  int lane, int wave) {
    if (wave < NT) {
        int nt = wave;
        v8f acc = {};
        for (int kt = 0; kt < 8; ++kt) {
            v16h a = load_a_frag(sH, HDIM, kt, lane);
            v16h b = load_b_frag(W2, 8, nt, kt, lane);
            acc = wmma_f16(a, b, acc);
        }
        int ln = lane & 15, hi = lane >> 4;
        int n = nt * 16 + ln;
        float bias = (n < Nout) ? b2[n] : 0.0f;
        for (int r = 0; r < 8; ++r)
            sOut[(hi * 8 + r) * strideOut + n] = acc[r] + bias;
    }
}

// ---------------------------------------------------------------------------
// Weight packing: fp32 KxN row-major -> f16 WMMA B-fragment layout (padded),
// placed at column offset nOff inside the packed destination.
// ---------------------------------------------------------------------------
__global__ __launch_bounds__(256) void pack_weights_kernel(
    const float* __restrict__ W, _Float16* __restrict__ P,
    int K, int N, int KP, int NPs, int nOff) {
    int i = blockIdx.x * 256 + threadIdx.x;
    if (i >= KP * NPs) return;
    int k = i / NPs, nl = i % NPs;
    int ng = nl + nOff;
    int kt = k >> 5, hi = (k >> 3) & 1, e = (k & 7) + ((k >> 4) & 1) * 8;
    int nt = ng >> 4, ln = ng & 15;
    int KT = KP >> 5;
    int lane = hi * 16 + ln;
    float v = (k < K && nl < N) ? W[(size_t)k * N + nl] : 0.0f;
    P[(((size_t)nt * KT + kt) * 32 + lane) * 16 + e] = (_Float16)v;
}

__global__ void zero_f32_kernel(float* __restrict__ p, long long n) {
    long long i = (long long)blockIdx.x * blockDim.x + threadIdx.x;
    long long stride = (long long)gridDim.x * blockDim.x;
    for (; i < n; i += stride) p[i] = 0.0f;
}

// ---------------------------------------------------------------------------
// Edge kernel: one block = 32 edges. Fused layer-1 (142 -> 512 = edge|gate),
// then 12 layer-2 output-tile tasks spread over 8 waves, then gated scatter.
// ---------------------------------------------------------------------------
__global__ __launch_bounds__(256) void edge_gnn_kernel(
    const float* __restrict__ z_src, const float* __restrict__ z_tgt,
    const long long* __restrict__ e_src, const long long* __restrict__ e_tgt, int E,
    const _Float16* __restrict__ W1cat,
    const float* __restrict__ b1e, const float* __restrict__ b1w,
    const _Float16* __restrict__ W2e, const float* __restrict__ b2e,
    const _Float16* __restrict__ W2w, const float* __restrict__ b2w,
    float* __restrict__ agg) {
    __shared__ int s_src[EROWS], s_tgt[EROWS];
    __shared__ float s_feat[EROWS][8];
    __shared__ __align__(16) _Float16 sX[EROWS][KP_EDGE];
    __shared__ __align__(16) _Float16 sH[EROWS * H2];
    __shared__ float sOutW[EROWS][16];
    __shared__ float sOut[EROWS][NP_OUT];
    __shared__ float sW[EROWS];

    const int t = threadIdx.x, lane = t & 31, wave = t >> 5;
    const int ln = lane & 15, hi = lane >> 4;
    const int e0 = blockIdx.x * EROWS;

    if (t == 0 && e0 + EROWS < E) {              // gfx1250 global_prefetch_b8
        __builtin_prefetch(&e_src[e0 + EROWS], 0, 0);
        __builtin_prefetch(&e_tgt[e0 + EROWS], 0, 0);
    }

    if (t < EROWS) {
        int e = e0 + t;
        int s = 0, g = 0;
        if (e < E) { s = (int)e_src[e]; g = (int)e_tgt[e]; }
        s_src[t] = s; s_tgt[t] = g;
        const float* zs = z_src + (size_t)s * FDIM;
        const float* zt = z_tgt + (size_t)g * FDIM;
        float d0 = zs[0] - zt[0], d1 = zs[1] - zt[1], d2 = zs[2] - zt[2];
        float a0 = zs[3], a1 = zs[4], a2 = zs[5];
        float b0 = zt[3], b1v = zt[4], b2v = zt[5];
        float c0 = a1 * b2v - a2 * b1v;
        float c1 = a2 * b0  - a0 * b2v;
        float c2 = a0 * b1v - a1 * b0;
        s_feat[t][0] = d0; s_feat[t][1] = d1; s_feat[t][2] = d2;
        s_feat[t][3] = d0 * d0 + d1 * d1 + d2 * d2;
        s_feat[t][4] = c0; s_feat[t][5] = c1; s_feat[t][6] = c2;
        s_feat[t][7] = sqrtf(c0 * c0 + c1 * c1 + c2 * c2);
    }
    __syncthreads();

    for (int i = t; i < EROWS * KP_EDGE; i += 256) {
        int r = i / KP_EDGE, c = i % KP_EDGE;
        float v = 0.0f;
        if (e0 + r < E) {
            if (c < FDIM)              v = z_src[(size_t)s_src[r] * FDIM + c];
            else if (c < 2 * FDIM)     v = z_tgt[(size_t)s_tgt[r] * FDIM + (c - FDIM)];
            else if (c < 2 * FDIM + 8) v = s_feat[r][c - 2 * FDIM];
        }
        sX[r][c] = (_Float16)v;
    }
    __syncthreads();

    // ---- fused layer 1: [32 x 160] @ [160 x 512] ; each wave owns 4 nt ----
    {
        const int KT = KP_EDGE / 32;
        v8f acc[2][4] = {};
        for (int kt = 0; kt < KT; ++kt) {
            v16h a0 = load_a_frag(&sX[0][0],  KP_EDGE, kt, lane);   // rows 0..15
            v16h a1 = load_a_frag(&sX[16][0], KP_EDGE, kt, lane);   // rows 16..31
            for (int j = 0; j < 4; ++j) {
                v16h b = load_b_frag(W1cat, KT, wave * 4 + j, kt, lane);
                acc[0][j] = wmma_f16(a0, b, acc[0][j]);
                acc[1][j] = wmma_f16(a1, b, acc[1][j]);
            }
        }
        for (int j = 0; j < 4; ++j) {
            int n = (wave * 4 + j) * 16 + ln;
            float bias = (n < HDIM) ? b1e[n] : b1w[n - HDIM];
            for (int rt = 0; rt < 2; ++rt)
                for (int r = 0; r < 8; ++r) {
                    float v = acc[rt][j][r] + bias;
                    sH[(rt * 16 + hi * 8 + r) * H2 + n] = (_Float16)fmaxf(v, 0.0f);
                }
        }
    }
    __syncthreads();

    // ---- layer 2: 10 edge-out tiles (2 rt x 5 nt) + 2 gate tiles, 8 waves ----
    for (int task = wave; task < 12; task += 8) {
        int isw = task >= 10;
        int rt  = isw ? (task - 10) : (task / 5);
        int nt  = isw ? 0 : (task % 5);
        const _Float16* hbase = sH + (size_t)rt * 16 * H2 + (isw ? HDIM : 0);
        v8f acc = {};
        for (int kt = 0; kt < 8; ++kt) {
            v16h a = load_a_frag(hbase, H2, kt, lane);
            v16h b = load_b_frag(isw ? W2w : W2e, 8, nt, kt, lane);
            acc = wmma_f16(a, b, acc);
        }
        int n = nt * 16 + ln;
        if (isw) {
            float bias = (n == 0) ? b2w[0] : 0.0f;
            for (int r = 0; r < 8; ++r)
                sOutW[rt * 16 + hi * 8 + r][n] = acc[r] + bias;
        } else {
            float bias = (n < FDIM) ? b2e[n] : 0.0f;
            for (int r = 0; r < 8; ++r)
                sOut[rt * 16 + hi * 8 + r][n] = acc[r] + bias;
        }
    }
    __syncthreads();

    if (t < EROWS) sW[t] = 1.0f / (1.0f + expf(-sOutW[t][0]));
    __syncthreads();

    for (int i = t; i < EROWS * FDIM; i += 256) {
        int r = i / FDIM, c = i % FDIM;
        if (e0 + r < E)
            atomicAdd(&agg[(size_t)s_tgt[r] * FDIM + c], sOut[r][c] * sW[r]);
    }
}

// ---------------------------------------------------------------------------
// World kernel: per-node 70 -> 256 -> 67 MLP, sum over nodes into wsum[67].
// ---------------------------------------------------------------------------
__global__ __launch_bounds__(256) void world_gnn_kernel(
    const float* __restrict__ z, int N,
    const _Float16* __restrict__ W1, const float* __restrict__ b1,
    const _Float16* __restrict__ W2, const float* __restrict__ b2,
    const float* __restrict__ pos_world,
    float* __restrict__ wsum) {
    __shared__ __align__(16) _Float16 sX[16][KP_WORLD];
    __shared__ __align__(16) _Float16 sH[16 * HDIM];
    __shared__ float sOut[16][NP_OUT];

    const int t = threadIdx.x, lane = t & 31, wave = t >> 5;
    const int n0 = blockIdx.x * 16;

    for (int i = t; i < 16 * KP_WORLD; i += 256) {
        int r = i / KP_WORLD, c = i % KP_WORLD;
        float v = 0.0f;
        int node = n0 + r;
        if (node < N) {
            if (c < FDIM)          v = z[(size_t)node * FDIM + c];
            else if (c < FDIM + 3) v = z[(size_t)node * FDIM + (c - FDIM)] - pos_world[c - FDIM];
        }
        sX[r][c] = (_Float16)v;
    }
    __syncthreads();
    mlp_layer1(&sX[0][0], KP_WORLD, KP_WORLD / 32, W1, b1, sH, lane, wave);
    __syncthreads();
    mlp_layer2(sH, W2, b2, &sOut[0][0], NP_OUT, 5, FDIM, lane, wave);
    __syncthreads();

    int nrows = N - n0; if (nrows > 16) nrows = 16;
    if (t < FDIM) {
        float s = 0.0f;
        for (int r = 0; r < nrows; ++r) s += sOut[r][t];
        atomicAdd(&wsum[t], s);
    }
}

// ---------------------------------------------------------------------------
// Node kernel: [z | agg(+wsum)] 134 -> 256 -> 67 ; out[node] += result.
// Plain += is safe: kernels are serialized on the stream.
// ---------------------------------------------------------------------------
__global__ __launch_bounds__(256) void node_gnn_kernel(
    const float* __restrict__ z, const float* __restrict__ agg,
    const float* __restrict__ wsum, int has_wsum, int N,
    const _Float16* __restrict__ W1, const float* __restrict__ b1,
    const _Float16* __restrict__ W2, const float* __restrict__ b2,
    float* __restrict__ out) {
    __shared__ __align__(16) _Float16 sX[16][KP_NODE];
    __shared__ __align__(16) _Float16 sH[16 * HDIM];
    __shared__ float sOut[16][NP_OUT];

    const int t = threadIdx.x, lane = t & 31, wave = t >> 5;
    const int n0 = blockIdx.x * 16;

    for (int i = t; i < 16 * KP_NODE; i += 256) {
        int r = i / KP_NODE, c = i % KP_NODE;
        float v = 0.0f;
        int node = n0 + r;
        if (node < N) {
            if (c < FDIM)            v = z[(size_t)node * FDIM + c];
            else if (c < 2 * FDIM) {
                int cc = c - FDIM;
                v = agg[(size_t)node * FDIM + cc];
                if (has_wsum) v += wsum[cc];
            }
        }
        sX[r][c] = (_Float16)v;
    }
    __syncthreads();
    mlp_layer1(&sX[0][0], KP_NODE, KP_NODE / 32, W1, b1, sH, lane, wave);
    __syncthreads();
    mlp_layer2(sH, W2, b2, &sOut[0][0], NP_OUT, 5, FDIM, lane, wave);
    __syncthreads();

    for (int i = t; i < 16 * FDIM; i += 256) {
        int r = i / FDIM, c = i % FDIM;
        int node = n0 + r;
        if (node < N)
            out[(size_t)node * FDIM + c] += sOut[r][c];
    }
}

// ---------------------------------------------------------------------------
// Host side
// ---------------------------------------------------------------------------
extern "C" void kernel_launch(void* const* d_in, const int* in_sizes, int n_in,
                              void* d_out, int out_size, void* d_ws, size_t ws_size,
                              hipStream_t stream) {
    (void)n_in; (void)ws_size; (void)out_size;
    const float*     z_h  = (const float*)d_in[0];
    const float*     z_l  = (const float*)d_in[1];
    const long long* e_hh = (const long long*)d_in[2];
    const long long* e_ll = (const long long*)d_in[3];
    const long long* e_hl = (const long long*)d_in[4];
    const long long* e_lh = (const long long*)d_in[5];
    const float*     posw = (const float*)d_in[6];

    const int N_H  = in_sizes[0] / FDIM;
    const int N_L  = in_sizes[1] / FDIM;
    const int E_HH = in_sizes[2] / 2;
    const int E_LL = in_sizes[3] / 2;
    const int E_HL = in_sizes[4] / 2;
    const int E_LH = in_sizes[5] / 2;

    // params flattened in sorted-key pytree order:
    //   gnn_h(16) | gnn_h_l(12) | gnn_l(16) | gnn_l_h(12)
    //   per MLP: b1,b2,w1,w2 ; self: edge,node,weight,world ; cross: edge,node,weight
    const int P = 7;
    const int GH = P, GHL = P + 16, GL = P + 28, GLH = P + 44;
    auto f32 = [&](int i) { return (const float*)d_in[i]; };

    char*  ws  = (char*)d_ws;
    size_t off = 0;
    auto alloc = [&](size_t bytes) -> void* {
        off = (off + 255) & ~(size_t)255;
        void* p = ws + off;
        off += bytes;
        return p;
    };
    auto pack_into = [&](const float* W, _Float16* Pk, int K, int N, int KPp, int NPs, int nOff) {
        int tot = KPp * NPs;
        pack_weights_kernel<<<(tot + 255) / 256, 256, 0, stream>>>(W, Pk, K, N, KPp, NPs, nOff);
    };
    auto pack = [&](const float* W, int K, int N, int KPp, int NPp) -> _Float16* {
        _Float16* Pk = (_Float16*)alloc((size_t)KPp * NPp * sizeof(_Float16));
        pack_into(W, Pk, K, N, KPp, NPp, 0);
        return Pk;
    };

    struct GnnW {
        _Float16 *e1cat, *e2, *w2, *n1, *n2, *o1, *o2;
        const float *be1, *be2, *bw1, *bw2, *bn1, *bn2, *bo1, *bo2;
    };
    auto pack_gnn = [&](int base, bool has_world) -> GnnW {
        GnnW g{};
        // fused layer-1: edge w1 -> cols 0..255, weight w1 -> cols 256..511
        g.e1cat = (_Float16*)alloc((size_t)KP_EDGE * H2 * sizeof(_Float16));
        g.be1 = f32(base + 0); g.be2 = f32(base + 1);
        pack_into(f32(base + 2),  g.e1cat, 2 * FDIM + 8, HDIM, KP_EDGE, HDIM, 0);
        g.e2 = pack(f32(base + 3), HDIM, FDIM, HDIM, NP_OUT);
        // node
        g.bn1 = f32(base + 4); g.bn2 = f32(base + 5);
        g.n1 = pack(f32(base + 6), 2 * FDIM, HDIM, KP_NODE, HDIM);
        g.n2 = pack(f32(base + 7), HDIM, FDIM, HDIM, NP_OUT);
        // weight (gate)
        g.bw1 = f32(base + 8); g.bw2 = f32(base + 9);
        pack_into(f32(base + 10), g.e1cat, 2 * FDIM + 8, HDIM, KP_EDGE, HDIM, HDIM);
        g.w2 = pack(f32(base + 11), HDIM, 1, HDIM, 16);
        if (has_world) {
            g.bo1 = f32(base + 12); g.bo2 = f32(base + 13);
            g.o1 = pack(f32(base + 14), FDIM + 3, HDIM, KP_WORLD, HDIM);
            g.o2 = pack(f32(base + 15), HDIM, FDIM, HDIM, NP_OUT);
        }
        return g;
    };
    GnnW gh  = pack_gnn(GH,  true);
    GnnW ghl = pack_gnn(GHL, false);
    GnnW gl  = pack_gnn(GL,  true);
    GnnW glh = pack_gnn(GLH, false);

    // ---- scratch accumulators ----
    float* agg_h  = (float*)alloc((size_t)N_H * FDIM * sizeof(float));
    float* agg_l  = (float*)alloc((size_t)N_L * FDIM * sizeof(float));
    float* agg_hl = (float*)alloc((size_t)N_L * FDIM * sizeof(float));
    float* agg_lh = (float*)alloc((size_t)N_H * FDIM * sizeof(float));
    float* wsum_h = (float*)alloc(FDIM * sizeof(float));
    float* wsum_l = (float*)alloc(FDIM * sizeof(float));

    float* out_h = (float*)d_out;
    float* out_l = out_h + (size_t)N_H * FDIM;

    zero_f32_kernel<<<2048, 256, 0, stream>>>(agg_h,  (long long)N_H * FDIM);
    zero_f32_kernel<<<2048, 256, 0, stream>>>(agg_l,  (long long)N_L * FDIM);
    zero_f32_kernel<<<2048, 256, 0, stream>>>(agg_hl, (long long)N_L * FDIM);
    zero_f32_kernel<<<2048, 256, 0, stream>>>(agg_lh, (long long)N_H * FDIM);
    zero_f32_kernel<<<1,    256, 0, stream>>>(wsum_h, FDIM);
    zero_f32_kernel<<<1,    256, 0, stream>>>(wsum_l, FDIM);
    zero_f32_kernel<<<2048, 256, 0, stream>>>(out_h,  (long long)(N_H + N_L) * FDIM);

    // ---- edge message passing (fused WMMA MLPs + gated scatter) ----
    auto edges = [&](const float* zs, const float* zt, const long long* ei, int E,
                     const GnnW& g, float* agg) {
        edge_gnn_kernel<<<(E + EROWS - 1) / EROWS, 256, 0, stream>>>(
            zs, zt, ei, ei + E, E,
            g.e1cat, g.be1, g.bw1,
            g.e2, g.be2, g.w2, g.bw2, agg);
    };
    edges(z_h, z_h, e_hh, E_HH, gh,  agg_h);
    edges(z_l, z_l, e_ll, E_LL, gl,  agg_l);
    edges(z_h, z_l, e_hl, E_HL, ghl, agg_hl);
    edges(z_l, z_h, e_lh, E_LH, glh, agg_lh);

    // ---- world MLP sums (self GNNs only) ----
    world_gnn_kernel<<<(N_H + 15) / 16, 256, 0, stream>>>(
        z_h, N_H, gh.o1, gh.bo1, gh.o2, gh.bo2, posw, wsum_h);
    world_gnn_kernel<<<(N_L + 15) / 16, 256, 0, stream>>>(
        z_l, N_L, gl.o1, gl.bo1, gl.o2, gl.bo2, posw, wsum_l);

    // ---- node MLPs (stream-ordered += into d_out) ----
    node_gnn_kernel<<<(N_H + 15) / 16, 256, 0, stream>>>(
        z_h, agg_h, wsum_h, 1, N_H, gh.n1, gh.bn1, gh.n2, gh.bn2, out_h);
    node_gnn_kernel<<<(N_H + 15) / 16, 256, 0, stream>>>(
        z_h, agg_lh, wsum_h, 0, N_H, glh.n1, glh.bn1, glh.n2, glh.bn2, out_h);
    node_gnn_kernel<<<(N_L + 15) / 16, 256, 0, stream>>>(
        z_l, agg_l, wsum_l, 1, N_L, gl.n1, gl.bn1, gl.n2, gl.bn2, out_l);
    node_gnn_kernel<<<(N_L + 15) / 16, 256, 0, stream>>>(
        z_l, agg_hl, wsum_l, 0, N_L, ghl.n1, ghl.bn1, ghl.n2, ghl.bn2, out_l);
}
